// LDConv_3D_35218731827369
// MI455X (gfx1250) — compile-verified, matrix operand order checked by reading, not run on hardware
//
#include <hip/hip_runtime.h>
#include <math.h>

typedef __bf16 bf16_t;
typedef bf16_t v2bf  __attribute__((ext_vector_type(2)));
typedef bf16_t v16bf __attribute__((ext_vector_type(16)));
typedef float  v8f   __attribute__((ext_vector_type(8)));

#define BB     2
#define INC    64
#define OUTC   128
#define NP     8
#define DDIM   12
#define HDIM   48
#define WDIM   48
#define HW     (HDIM*WDIM)        // 2304
#define DHW    (DDIM*HW)          // 27648
#define M_TOT  (BB*DHW)           // 55296
#define KDIM   (INC*NP)           // 512
#define K1DIM  (27*INC)           // 1728 (stage-1 K, ordered k = tap*64 + ci)
#define BN_EPS 1e-5f

// ---------------------------------------------------------------------------
// Kernel 1a: pre-swizzle w_p into bf16 B-fragments for the stage-1 WMMA GEMM.
// Layout [kt(54)][ntile(2)][lane(32)][elem(16)], B[k', ch] with
// k' = tap*64 + ci, ch padded 24 -> 32 with zeros.
// ---------------------------------------------------------------------------
__global__ void k_w1prep(const float* __restrict__ w_p,
                         bf16_t* __restrict__ wb1) {
    int t     = blockIdx.x * 256 + threadIdx.x;     // 0 .. 55295
    int e     = t & 15;
    int lane  = (t >> 4) & 31;
    int ntile = (t >> 9) & 1;
    int kt    = t >> 10;                            // 0..53
    int K     = ((e < 8) ? e : (e + 8)) + (lane >> 4) * 8;
    int kp    = kt * 32 + K;
    int tap   = kp >> 6;
    int ci    = kp & 63;
    int ch    = ntile * 16 + (lane & 15);
    float v   = (ch < 24) ? w_p[ch * K1DIM + ci * 27 + tap] : 0.f;
    wb1[t] = (bf16_t)v;
}

// ---------------------------------------------------------------------------
// Kernel 1b: stage-1 offset conv as fused im2col + WMMA GEMM.
// M-tile = 64 positions/block; K = 1728 in 54 chunks of 32 (one conv tap per
// two chunks); double-buffered 4 KB A-fragment in LDS; 8 waves = 4 Mx2 N tiles.
// Fill path: batched branchless loads + packed v2bf LDS stores.
// ---------------------------------------------------------------------------
__global__ void k_offset_wmma(const float* __restrict__ x,
                              const bf16_t* __restrict__ wb1,
                              const float* __restrict__ b_p,
                              float* __restrict__ off) {
    __shared__ int    s_sp[64 * 27];       // spatial index per (ml, tap); -1 = OOB
    __shared__ bf16_t s_f[2][4 * 32 * 16]; // double-buffered A fragments (2 x 4 KB)

    const int tid   = threadIdx.x;
    const int m0    = blockIdx.x * 64;
    const int bb    = (m0 >= DHW) ? 1 : 0;     // block-uniform batch index
    const int sbase = m0 - bb * DHW;

    // ---- Phase A: im2col spatial-index table ----
    for (int j = tid; j < 64 * 27; j += 256) {
        int ml = j / 27, tap = j - ml * 27;
        int s  = sbase + ml;
        int d  = s / HW;
        int r  = s - d * HW;
        int h  = r / WDIM;
        int w  = r - h * WDIM;
        int dz = tap / 9, rem = tap - dz * 9, dx = rem / 3, dy = rem - dx * 3;
        int z = d + dz - 1, xx = h + dx - 1, yy = w + dy - 1;
        int sp = -1;
        if ((unsigned)z < DDIM && (unsigned)xx < HDIM && (unsigned)yy < WDIM)
            sp = (z * HDIM + xx) * WDIM + yy;
        s_sp[j] = sp;
    }

    const int mlb = tid & 31;     // lane sweeps positions -> coalesced x loads
    const int kb  = tid >> 5;     // 0..7 -> K pairs {2kb,2kb+1} and {+16}
    const float* xb = x + (size_t)bb * INC * DHW;

    // Per-thread constant fragment addressing
    const int khalf = (kb >> 2) & 1;           // of kk0 = 2*kb
    const int e0    = (kb << 1) & 7;
    const int lanew = (mlb & 15) + (khalf << 4);
    const int mt0   = mlb >> 4;                // ml=mlb   -> mtile mt0
                                               // ml=mlb+32-> mtile mt0+2
    const int fb0   = (mt0 * 32 + lanew) * 16;
    const int fb1   = ((mt0 + 2) * 32 + lanew) * 16;

    auto fill = [&](int kt, int buf) {
        int tap = kt >> 1;
        int cib = ((kt & 1) << 5) + (kb << 1);
        int sp0 = s_sp[mlb * 27 + tap];
        int sp1 = s_sp[(mlb + 32) * 27 + tap];
        int p0  = (sp0 < 0) ? 0 : sp0;
        int p1  = (sp1 < 0) ? 0 : sp1;
        // batch all 8 loads (one clause / one wait)
        float v00 = xb[(cib + 0)  * DHW + p0];
        float v01 = xb[(cib + 1)  * DHW + p0];
        float v02 = xb[(cib + 16) * DHW + p0];
        float v03 = xb[(cib + 17) * DHW + p0];
        float v10 = xb[(cib + 0)  * DHW + p1];
        float v11 = xb[(cib + 1)  * DHW + p1];
        float v12 = xb[(cib + 16) * DHW + p1];
        float v13 = xb[(cib + 17) * DHW + p1];
        if (sp0 < 0) { v00 = 0.f; v01 = 0.f; v02 = 0.f; v03 = 0.f; }
        if (sp1 < 0) { v10 = 0.f; v11 = 0.f; v12 = 0.f; v13 = 0.f; }
        bf16_t* f = &s_f[buf][0];
        *(v2bf*)&f[fb0 + e0]     = v2bf{(bf16_t)v00, (bf16_t)v01};
        *(v2bf*)&f[fb0 + e0 + 8] = v2bf{(bf16_t)v02, (bf16_t)v03};
        *(v2bf*)&f[fb1 + e0]     = v2bf{(bf16_t)v10, (bf16_t)v11};
        *(v2bf*)&f[fb1 + e0 + 8] = v2bf{(bf16_t)v12, (bf16_t)v13};
    };

    __syncthreads();          // s_sp ready
    fill(0, 0);
    __syncthreads();

    const int wv = tid >> 5, lane = tid & 31;
    const int mtile = wv & 3, ntile = wv >> 2;   // 4 x 2 output subtiles

    v8f acc = {};
    for (int kt = 0; kt < 54; ++kt) {
        int buf = kt & 1;
        v16bf a = *(const v16bf*)(&s_f[buf][(mtile * 32 + lane) * 16]);
        v16bf b = *(const v16bf*)(wb1 + ((kt * 2 + ntile) * 32 + lane) * 16);
        acc = __builtin_amdgcn_wmma_f32_16x16x32_bf16(
            false, a, false, b, (short)0, acc, false, false);
        if (kt + 1 < 54) fill(kt + 1, buf ^ 1);
        __syncthreads();
    }

    // Epilogue: add bias, drop padded channels 24..31, store offsets.
    int rbase = (lane >> 4) * 8;
    int col   = lane & 15;
    int ch    = ntile * 16 + col;
    if (ch < 24) {
        float bias = b_p[ch];
        float* op  = off + (bb * 24 + ch) * DHW + sbase + mtile * 16 + rbase;
        #pragma unroll
        for (int v = 0; v < 8; ++v)
            op[v] = acc[v] + bias;
    }
}

// ---------------------------------------------------------------------------
// Kernel 2: pre-swizzle w_conv ([o][k] flat, k=c*8+n) into bf16 B-fragments
// [kt(16)][ntile(8)][lane(32)][elem(16)] for the stage-3 WMMA GEMM.
// ---------------------------------------------------------------------------
__global__ void k_wprep(const float* __restrict__ w_conv,
                        bf16_t* __restrict__ wb) {
    int t     = blockIdx.x * 256 + threadIdx.x;     // 0 .. 65535
    int e     = t & 15;
    int lane  = (t >> 4) & 31;
    int ntile = (t >> 9) & 7;
    int kt    = t >> 12;
    int K     = ((e < 8) ? e : (e + 8)) + (lane >> 4) * 8;
    int k     = kt * 32 + K;
    int n     = ntile * 16 + (lane & 15);
    wb[t] = (bf16_t)w_conv[n * KDIM + k];
}

// ---------------------------------------------------------------------------
// Kernel 3 (fused): deformable gather -> LDS A-fragments -> WMMA GEMM.
// 864 blocks x 256 threads; each block owns a 64-position M-tile and the
// full N=128 output, K=512 (c*8+n).
// ---------------------------------------------------------------------------
__global__ void k_deform_gemm(const float* __restrict__ x,
                              const float* __restrict__ off,
                              const bf16_t* __restrict__ wb,
                              float* __restrict__ out) {
    extern __shared__ char smem[];
    bf16_t* s_a  = (bf16_t*)smem;                   // [4][16][32][16] bf16 = 64 KB
    int*    s_i0 = (int*)(smem + 65536);            // SoA [n][ml], 512 each
    int*    s_i1 = s_i0 + 512;
    int*    s_i2 = s_i0 + 1024;
    int*    s_i3 = s_i0 + 1536;
    float*  s_w0 = (float*)(smem + 65536 + 8192);
    float*  s_w1 = s_w0 + 512;
    float*  s_w2 = s_w0 + 1024;
    float*  s_w3 = s_w0 + 1536;

    const int tid   = threadIdx.x;
    const int m0    = blockIdx.x * 64;
    const int bb    = (m0 >= DHW) ? 1 : 0;     // block-uniform batch index
    const int sbase = m0 - bb * DHW;

    // ---- Phase A: per (n, ml) corner indices + weights -> LDS (SoA) ----
    for (int j = tid; j < 512; j += 256) {
        int n  = j >> 6, ml = j & 63;
        int s  = sbase + ml;
        int d  = s / HW;
        int r  = s - d * HW;
        int h  = r / WDIM;
        int w  = r - h * WDIM;

        const float* op = off + (bb * 24) * DHW + s;
        float oz = op[n * DHW];
        float ox = op[(8 + n) * DHW];
        float oy = op[(16 + n) * DHW];

        // _p_n(8): z={0,0,0,1,1,1,2,2} x={0,0,0,0,0,0,1,1} y={0,1,2,0,1,2,0,1}
        float pnz = (n < 6) ? (float)(n / 3) : 2.f;
        float pnx = (n < 6) ? 0.f : 1.f;
        float pny = (n < 6) ? (float)(n % 3) : (float)(n - 6);

        float pz = oz + (float)d + pnz;
        float px = ox + (float)h + pnx;
        float py = oy + (float)w + pny;
        float fz = floorf(pz), fx = floorf(px), fy = floorf(py);

        float qltz = fminf(fmaxf(fz,       0.f), DDIM - 1.f);
        float qltx = fminf(fmaxf(fx,       0.f), HDIM - 1.f);
        float qlty = fminf(fmaxf(fy,       0.f), WDIM - 1.f);
        float qrbz = fminf(fmaxf(fz + 1.f, 0.f), DDIM - 1.f);
        float qrbx = fminf(fmaxf(fx + 1.f, 0.f), HDIM - 1.f);
        float qrby = fminf(fmaxf(fy + 1.f, 0.f), WDIM - 1.f);
        float pzc  = fminf(fmaxf(pz, 0.f), DDIM - 1.f);
        float pxc  = fminf(fmaxf(px, 0.f), HDIM - 1.f);
        float pyc  = fminf(fmaxf(py, 0.f), WDIM - 1.f);

        float az = 1.f + qltz - pzc, bz = 1.f - qrbz + pzc;
        float ax = 1.f + qltx - pxc, bx = 1.f - qrbx + pxc;
        float ay = 1.f + qlty - pyc, by = 1.f - qrby + pyc;

        s_w0[j] = az * ax * ay;   // g_lt
        s_w1[j] = bz * bx * by;   // g_rb
        s_w2[j] = az * bx * ay;   // g_lb
        s_w3[j] = bz * ax * by;   // g_rt

        int z0 = (int)qltz, x0 = (int)qltx, y0 = (int)qlty;
        int z1 = (int)qrbz, x1 = (int)qrbx, y1 = (int)qrby;
        s_i0[j] = (z0 * HDIM + x0) * WDIM + y0;  // lt
        s_i1[j] = (z1 * HDIM + x1) * WDIM + y1;  // rb
        s_i2[j] = (z0 * HDIM + x1) * WDIM + y0;  // lb
        s_i3[j] = (z1 * HDIM + x0) * WDIM + y1;  // rt
    }
    __syncthreads();

    // ---- Phase B: gather + blend -> bf16 A-fragments in LDS ----
    // Wave-uniform k-pair (same channel slab), lanes sweep positions;
    // 8 gathers batched per iteration, packed v2bf LDS stores.
    {
        const int mlb = tid & 31;
        const int wv  = tid >> 5;
        const float* xbase = x + (size_t)bb * INC * DHW;
        #pragma unroll 2
        for (int it = 0; it < 64; ++it) {
            int ml = mlb + ((it & 1) << 5);
            int kp = wv * 32 + (it >> 1);       // uniform within the wave
            int k0 = kp << 1;                   // even -> k0,k0+1 share channel
            int c  = k0 >> 3;
            int n0 = k0 & 7;
            int jj0 = n0 * 64 + ml;
            int jj1 = jj0 + 64;
            const float* xc = xbase + c * DHW;
            int a0 = s_i0[jj0], a1 = s_i1[jj0], a2 = s_i2[jj0], a3 = s_i3[jj0];
            int b0 = s_i0[jj1], b1 = s_i1[jj1], b2 = s_i2[jj1], b3 = s_i3[jj1];
            float xa0 = xc[a0], xa1 = xc[a1], xa2 = xc[a2], xa3 = xc[a3];
            float xb0 = xc[b0], xb1 = xc[b1], xb2 = xc[b2], xb3 = xc[b3];
            float va = s_w0[jj0] * xa0 + s_w1[jj0] * xa1
                     + s_w2[jj0] * xa2 + s_w3[jj0] * xa3;
            float vb = s_w0[jj1] * xb0 + s_w1[jj1] * xb1
                     + s_w2[jj1] * xb2 + s_w3[jj1] * xb3;
            int kk    = k0 & 31;
            int kt    = k0 >> 5;
            int lanew = (ml & 15) + ((kk & 8) << 1);
            int e     = (kk & 7) | ((kk & 16) >> 1);   // even
            *(v2bf*)&s_a[(((ml >> 4) * 16 + kt) * 32 + lanew) * 16 + e] =
                v2bf{(bf16_t)va, (bf16_t)vb};
        }
    }
    __syncthreads();

    // ---- Phase C: 8 waves x 4 WMMA output subtiles each (64x128 total) ----
    {
        int wv    = tid >> 5;
        int lane  = tid & 31;
        int mtile = wv & 3;
        int nquad = wv >> 2;

        v8f acc[4] = {};
        #pragma unroll
        for (int kt = 0; kt < 16; ++kt) {
            v16bf a = *(const v16bf*)(s_a + ((mtile * 16 + kt) * 32 + lane) * 16);
            #pragma unroll
            for (int j = 0; j < 4; ++j) {
                v16bf bfr = *(const v16bf*)(wb + ((kt * 8 + nquad * 4 + j) * 32 + lane) * 16);
                acc[j] = __builtin_amdgcn_wmma_f32_16x16x32_bf16(
                    false, a, false, bfr, (short)0, acc[j], false, false);
            }
        }

        // D layout: VGPR v -> M = v + 8*(lane/16), N = lane%16
        int rbase = (lane >> 4) * 8;
        int col   = lane & 15;
        #pragma unroll
        for (int j = 0; j < 4; ++j) {
            int o = (nquad * 4 + j) * 16 + col;
            float* op = out + (bb * OUTC + o) * DHW + sbase + mtile * 16 + rbase;
            #pragma unroll
            for (int v = 0; v < 8; ++v)
                op[v] = acc[j][v];
        }
    }
}

// ---------------------------------------------------------------------------
// Kernel 4: deterministic per-channel BN stats (one block per channel).
// ---------------------------------------------------------------------------
__global__ void k_bn_stats(const float* __restrict__ out,
                           float* __restrict__ stats) {
    __shared__ float rs[256], rq[256];
    int o = blockIdx.x, tid = threadIdx.x;
    float s = 0.f, q = 0.f;
    for (int b = 0; b < BB; ++b) {
        const float* p = out + (b * OUTC + o) * DHW;
        for (int i = tid; i < DHW; i += 256) {
            float v = p[i];
            s += v;
            q = fmaf(v, v, q);
        }
    }
    rs[tid] = s; rq[tid] = q;
    __syncthreads();
    for (int st = 128; st > 0; st >>= 1) {
        if (tid < st) { rs[tid] += rs[tid + st]; rq[tid] += rq[tid + st]; }
        __syncthreads();
    }
    if (tid == 0) {
        float inv  = 1.f / (float)(BB * DHW);
        float mean = rs[0] * inv;
        float var  = rq[0] * inv - mean * mean;
        stats[o]        = mean;
        stats[OUTC + o] = rsqrtf(var + BN_EPS);
    }
}

// ---------------------------------------------------------------------------
// Kernel 5: fused BN apply + SiLU, in place on d_out.
// ---------------------------------------------------------------------------
__global__ void k_bn_silu(float* __restrict__ out,
                          const float* __restrict__ stats,
                          const float* __restrict__ gamma,
                          const float* __restrict__ beta) {
    int i = blockIdx.x * 256 + threadIdx.x;
    int o = (i / DHW) & (OUTC - 1);
    float v = (out[i] - stats[o]) * stats[OUTC + o] * gamma[o] + beta[o];
    out[i] = v / (1.f + __expf(-v));
}

// ---------------------------------------------------------------------------
extern "C" void kernel_launch(void* const* d_in, const int* in_sizes, int n_in,
                              void* d_out, int out_size, void* d_ws, size_t ws_size,
                              hipStream_t stream) {
    const float* x      = (const float*)d_in[0];
    const float* w_p    = (const float*)d_in[1];
    const float* b_p    = (const float*)d_in[2];
    const float* w_conv = (const float*)d_in[3];
    const float* gamma  = (const float*)d_in[4];
    const float* beta   = (const float*)d_in[5];
    float* out = (float*)d_out;

    // Workspace carve-up (~5.55 MB total)
    float*  ws_off   = (float*)d_ws;                                    // BB*24*DHW f32
    bf16_t* ws_b     = (bf16_t*)((char*)d_ws + (size_t)(BB*24*DHW)*4);  // 512*128 bf16
    float*  ws_stats = (float*)((char*)ws_b + (size_t)KDIM*OUTC*2);     // 256 f32
    bf16_t* ws_b1    = (bf16_t*)((char*)ws_stats + 1024);               // 54*2*512 bf16

    k_w1prep<<<(54 * 2 * KDIM) / 256, 256, 0, stream>>>(w_p, ws_b1);
    k_wprep<<<(KDIM * OUTC) / 256, 256, 0, stream>>>(w_conv, ws_b);
    k_offset_wmma<<<M_TOT / 64, 256, 0, stream>>>(x, ws_b1, b_p, ws_off);
    k_deform_gemm<<<M_TOT / 64, 256, 81920, stream>>>(x, ws_off, ws_b, out);
    k_bn_stats<<<OUTC, 256, 0, stream>>>(out, ws_stats);
    k_bn_silu<<<(BB * OUTC * DHW) / 256, 256, 0, stream>>>(out, ws_stats, gamma, beta);
}
